// VllmMixtureOfExpertsOpFP8_71141838291318
// MI455X (gfx1250) — compile-verified, matrix operand order checked by reading, not run on hardware
//
#include <hip/hip_runtime.h>
#include <hip/hip_bf16.h>

// ---- problem constants (match reference setup_inputs) ----
#define T_DIM 4096
#define H_DIM 2048
#define I_DIM 1792
#define E_NUM 8
#define TOPK  2
#define BM    32      // tokens per block = 2 WMMA m-tiles
#define QBS   128     // fp8 block-scale size

typedef __attribute__((ext_vector_type(16))) __bf16 v16bf;
typedef __attribute__((ext_vector_type(2)))  __bf16 v2bf;
typedef __attribute__((ext_vector_type(8)))  float  v8f;

union Frag {           // one WMMA 16x32 bf16 operand (8 VGPRs)
    v16bf    v;
    unsigned u[8];
    float4   f4[2];
};

// pack two f32 -> two bf16 in one dword: native V_CVT_PK_BF16_F32 (CDNA4/5)
__device__ __forceinline__ unsigned pk2(float lo, float hi) {
    unsigned r;
    asm("v_cvt_pk_bf16_f32 %0, %1, %2" : "=v"(r) : "v"(lo), "v"(hi));
    return r;
}

__device__ __forceinline__ __bf16 f2bf1(float f) {
    unsigned r = pk2(f, f);
    return __builtin_bit_cast(v2bf, r).x;
}

// Load A fragment from an LDS row (bf16).
// lane L: m = L&15, half = L>>4; elem i -> k = k0 + (i<8?0:16) + half*8 + i%8
__device__ __forceinline__ void fill_a(Frag& f, const __bf16* p) {
    f.f4[0] = *(const float4*)(p);        // k .. k+7   (16B)
    f.f4[1] = *(const float4*)(p + 16);   // k+16..k+23 (16B)
}

// Load B fragment from a global f32 row, convert to bf16 (NO scaling here:
// the block scale is folded into the accumulator once per 128-k block).
__device__ __forceinline__ void fill_b(Frag& f, const float* p) {
    float4 q0 = *(const float4*)(p);
    float4 q1 = *(const float4*)(p + 4);
    float4 q2 = *(const float4*)(p + 16);
    float4 q3 = *(const float4*)(p + 20);
    f.u[0] = pk2(q0.x, q0.y);
    f.u[1] = pk2(q0.z, q0.w);
    f.u[2] = pk2(q1.x, q1.y);
    f.u[3] = pk2(q1.z, q1.w);
    f.u[4] = pk2(q2.x, q2.y);
    f.u[5] = pk2(q2.z, q2.w);
    f.u[6] = pk2(q3.x, q3.y);
    f.u[7] = pk2(q3.z, q3.w);
}

// acc += s * tmp  (scale fold, once per 128-k block)
__device__ __forceinline__ void fold(v8f& c, const v8f& t, float s) {
    #pragma unroll
    for (int j = 0; j < 8; ++j) c[j] = __builtin_fmaf(s, t[j], c[j]);
}

#define WMMA_BF16(A, B, C) \
    __builtin_amdgcn_wmma_f32_16x16x32_bf16(false, (A), false, (B), (short)0, (C), false, false)

// ---------------- routing: sort (token, weight) by expert, pad to BM ----------------
__global__ __launch_bounds__(256) void moe_route_kernel(
    const int* __restrict__ topk_ids, const float* __restrict__ topk_w,
    int* __restrict__ blk_expert, int* __restrict__ sorted_tok,
    float* __restrict__ sorted_w, int maxBlocks)
{
    __shared__ int cnt[E_NUM];
    __shared__ int off[E_NUM];
    __shared__ int cur[E_NUM];
    const int tid = threadIdx.x;

    for (int i = tid; i < maxBlocks * BM; i += 256) { sorted_tok[i] = -1; sorted_w[i] = 0.f; }
    if (tid < E_NUM) { cnt[tid] = 0; cur[tid] = 0; }
    __syncthreads();

    for (int i = tid; i < T_DIM * TOPK; i += 256)
        atomicAdd(&cnt[topk_ids[i]], 1);
    __syncthreads();

    if (tid == 0) {
        int run = 0;
        for (int e = 0; e < E_NUM; ++e) {
            off[e] = run;
            int nb = (cnt[e] + BM - 1) / BM;
            for (int b = run / BM; b < run / BM + nb; ++b) blk_expert[b] = e;
            run += nb * BM;
        }
        for (int b = run / BM; b < maxBlocks; ++b) blk_expert[b] = -1;
    }
    __syncthreads();

    for (int i = tid; i < T_DIM * TOPK; i += 256) {
        int e = topk_ids[i];
        int pos = off[e] + atomicAdd(&cur[e], 1);
        sorted_tok[pos] = i / TOPK;
        sorted_w[pos]   = topk_w[i];
    }
}

// ---------------- fused grouped GEMM1 + SwiGLU + GEMM2 ----------------
__global__ __launch_bounds__(256) void moe_fused_kernel(
    const float* __restrict__ x,
    const float* __restrict__ w13, const float* __restrict__ w13_s,
    const float* __restrict__ w2,  const float* __restrict__ w2_s,
    const int* __restrict__ blk_expert, const int* __restrict__ sorted_tok,
    const float* __restrict__ sorted_w, float* __restrict__ out)
{
    __shared__ __align__(16) __bf16 xs[BM * H_DIM];     // 128 KB: gathered x rows (bf16)
    __shared__ __align__(16) __bf16 acts[BM * I_DIM];   // 112 KB: SwiGLU activations
    __shared__ int   tok_s[BM];
    __shared__ float w_s[BM];

    const int b = blockIdx.x;
    const int e = blk_expert[b];
    if (e < 0) return;

    const int tid  = threadIdx.x;
    const int wave = tid >> 5;
    const int lane = tid & 31;
    const int half = lane >> 4;
    const int ln   = lane & 15;

    if (tid < BM) {
        tok_s[tid] = sorted_tok[b * BM + tid];
        w_s[tid]   = sorted_w[b * BM + tid];
    }
    __syncthreads();

    // ---- stage x rows into LDS as bf16 (sentinel rows -> 0) ----
    for (int idx = tid; idx < BM * (H_DIM / 4); idx += 256) {
        int m  = idx >> 9;               // / (H_DIM/4)
        int k4 = (idx & 511) << 2;
        int tok = tok_s[m];
        unsigned lo = 0u, hi = 0u;
        if (tok >= 0) {
            float4 f = *(const float4*)(x + (size_t)tok * H_DIM + k4);
            lo = pk2(f.x, f.y);
            hi = pk2(f.z, f.w);
        }
        *(uint2*)(&xs[m * H_DIM + k4]) = make_uint2(lo, hi);
    }
    __syncthreads();

    // ---- GEMM1: h = x @ w13_e^T (gate+up), then SwiGLU -> acts (LDS) ----
    const float* w13sc = w13_s + (size_t)e * ((2 * I_DIM) / QBS) * (H_DIM / QBS);
    for (int n0 = 0; n0 < I_DIM; n0 += 128) {
        const int gcol = n0 + wave * 16;           // gate row block in w13
        const int ucol = I_DIM + gcol;             // up   row block in w13
        v8f cg0 = {}, cg1 = {}, cu0 = {}, cu1 = {};
        const float* wg_row = w13 + ((size_t)e * 2 * I_DIM + gcol + ln) * H_DIM + half * 8;
        const float* wu_row = w13 + ((size_t)e * 2 * I_DIM + ucol + ln) * H_DIM + half * 8;
        const float* sg_p = w13sc + (gcol / QBS) * (H_DIM / QBS);
        const float* su_p = w13sc + (ucol / QBS) * (H_DIM / QBS);
        const __bf16* arow0 = xs + ln * H_DIM + half * 8;
        const __bf16* arow1 = arow0 + 16 * H_DIM;

        for (int kb = 0; kb < H_DIM; kb += QBS) {
            const float sg = sg_p[kb / QBS];
            const float su = su_p[kb / QBS];
            __builtin_prefetch(wg_row + kb + QBS, 0, 0);
            __builtin_prefetch(wu_row + kb + QBS, 0, 0);
            v8f tg0 = {}, tg1 = {}, tu0 = {}, tu1 = {};   // unscaled partial sums
            #pragma unroll
            for (int kk = 0; kk < QBS; kk += 32) {
                const int k0 = kb + kk;
                Frag a0, a1, bg, bu;
                fill_a(a0, arow0 + k0);
                fill_a(a1, arow1 + k0);
                fill_b(bg, wg_row + k0);
                fill_b(bu, wu_row + k0);
                tg0 = WMMA_BF16(a0.v, bg.v, tg0);
                tg1 = WMMA_BF16(a1.v, bg.v, tg1);
                tu0 = WMMA_BF16(a0.v, bu.v, tu0);
                tu1 = WMMA_BF16(a1.v, bu.v, tu1);
            }
            fold(cg0, tg0, sg);
            fold(cg1, tg1, sg);
            fold(cu0, tu0, su);
            fold(cu1, tu1, su);
        }

        #pragma unroll
        for (int i = 0; i < 8; ++i) {
            // m-tile 0
            float g = cg0[i], u = cu0[i];
            float s = g * __builtin_amdgcn_rcpf(1.f + __expf(-g));   // fast silu
            acts[(i + 8 * half) * I_DIM + gcol + ln] = f2bf1(s * u);
            // m-tile 1
            g = cg1[i]; u = cu1[i];
            s = g * __builtin_amdgcn_rcpf(1.f + __expf(-g));
            acts[(16 + i + 8 * half) * I_DIM + gcol + ln] = f2bf1(s * u);
        }
    }
    __syncthreads();

    // ---- GEMM2: out += r * (acts @ w2_e^T) ----
    const float* w2sc = w2_s + (size_t)e * (H_DIM / QBS) * (I_DIM / QBS);
    for (int n0 = 0; n0 < H_DIM; n0 += 128) {
        const int col = n0 + wave * 16;
        v8f c0 = {}, c1 = {};
        const float* w2_row = w2 + ((size_t)e * H_DIM + col + ln) * I_DIM + half * 8;
        const float* sc_p = w2sc + (col / QBS) * (I_DIM / QBS);
        const __bf16* arow0 = acts + ln * I_DIM + half * 8;
        const __bf16* arow1 = arow0 + 16 * I_DIM;

        for (int kb = 0; kb < I_DIM; kb += QBS) {
            const float s = sc_p[kb / QBS];
            __builtin_prefetch(w2_row + kb + QBS, 0, 0);
            v8f t0 = {}, t1 = {};
            #pragma unroll
            for (int kk = 0; kk < QBS; kk += 32) {
                const int k0 = kb + kk;
                Frag a0, a1, bb;
                fill_a(a0, arow0 + k0);
                fill_a(a1, arow1 + k0);
                fill_b(bb, w2_row + k0);
                t0 = WMMA_BF16(a0.v, bb.v, t0);
                t1 = WMMA_BF16(a1.v, bb.v, t1);
            }
            fold(c0, t0, s);
            fold(c1, t1, s);
        }

        #pragma unroll
        for (int i = 0; i < 8; ++i) {
            int m0 = i + 8 * half;
            int m1 = 16 + m0;
            int tok0 = tok_s[m0];
            int tok1 = tok_s[m1];
            if (tok0 >= 0)
                unsafeAtomicAdd(out + (size_t)tok0 * H_DIM + col + ln, c0[i] * w_s[m0]);
            if (tok1 >= 0)
                unsafeAtomicAdd(out + (size_t)tok1 * H_DIM + col + ln, c1[i] * w_s[m1]);
        }
    }
}

extern "C" void kernel_launch(void* const* d_in, const int* in_sizes, int n_in,
                              void* d_out, int out_size, void* d_ws, size_t ws_size,
                              hipStream_t stream) {
    const float* x        = (const float*)d_in[0];
    const int*   topk_ids = (const int*)d_in[1];
    const float* topk_w   = (const float*)d_in[2];
    const float* w13      = (const float*)d_in[3];
    const float* w13_s    = (const float*)d_in[4];
    const float* w2       = (const float*)d_in[5];
    const float* w2_s     = (const float*)d_in[6];
    float* out = (float*)d_out;

    const int maxBlocks = (T_DIM * TOPK) / BM + E_NUM;   // 264

    // workspace layout (~68 KB)
    int*   blk_expert = (int*)d_ws;                      // maxBlocks
    int*   sorted_tok = blk_expert + maxBlocks;          // maxBlocks*BM
    float* sorted_w   = (float*)(sorted_tok + maxBlocks * BM);

    hipMemsetAsync(d_out, 0, sizeof(float) * (size_t)T_DIM * H_DIM, stream);

    moe_route_kernel<<<1, 256, 0, stream>>>(topk_ids, topk_w,
                                            blk_expert, sorted_tok, sorted_w, maxBlocks);

    moe_fused_kernel<<<maxBlocks, 256, 0, stream>>>(x, w13, w13_s, w2, w2_s,
                                                    blk_expert, sorted_tok, sorted_w, out);
}
